// BasisGATLayer_70995809403512
// MI455X (gfx1250) — matrix operand order repaired and checked by reading, not compiled, not run on hardware
//
#include <hip/hip_runtime.h>

// BasisGAT layer, MI455X (gfx1250, wave32, WMMA).
// Algebraic reductions: edge-MLP GEMM1 factorized into U[i]+V[j]; GEMM2 folded
// through `we` into 8 per-head vectors; q/k projections folded through wq/wk.
// Remaining GEMM-shaped work runs on v_wmma_f32_16x16x32_f16 (f32 accumulate).
// Workspace requirement: 811008 floats = ~3.1 MB.

typedef __attribute__((ext_vector_type(16))) _Float16 v16h;
typedef __attribute__((ext_vector_type(8)))  _Float16 v8h;
typedef __attribute__((ext_vector_type(8)))  float    v8f;

#define DD  256
#define HH  8
#define HDD 32
#define SS  255
#define SS1 256
#define BB  4

// workspace layout (float offsets)
#define WS_QWEFF 0            // [8][256]
#define WS_KWEFF 2048         // [8][256]
#define WS_W2EFF 4096         // [8][256]
#define WS_BIAS  6144         // [0..7] qbeff, [8..15] kbeff, [16..23] ebeff, [24] ab
#define WS_VPROJ 8192         // [1024][256]
#define WS_QK    270336       // [1024][16]  cols 0..7 = qlog, 8..15 = klog
#define WS_U     286720       // [4][256][256] (row r = desc row r; r=255 unused)
#define WS_V     548864       // [4][256][256]
// end = 811008 floats

__device__ __forceinline__ float wred_sum(float x) {
#pragma unroll
  for (int o = 16; o >= 1; o >>= 1) x += __shfl_xor(x, o, 32);
  return x;
}
__device__ __forceinline__ float wred_max(float x) {
#pragma unroll
  for (int o = 16; o >= 1; o >>= 1) x = fmaxf(x, __shfl_xor(x, o, 32));
  return x;
}

// ---------------- kernel 0: fold small weights -------------------------------
__global__ __launch_bounds__(256) void k0_precompute(
    const float* __restrict__ qW, const float* __restrict__ qb,
    const float* __restrict__ kW, const float* __restrict__ kbias,
    const float* __restrict__ eW2, const float* __restrict__ eb2,
    const float* __restrict__ aW, const float* __restrict__ ab,
    float* __restrict__ ws) {
  int tid = blockIdx.x * 256 + threadIdx.x;
  if (tid < 2048) {                       // qWeff[h,d] = sum_t qW[h*32+t, d] * wq[t]
    int h = tid >> 8, d = tid & 255;
    float s = 0.f;
#pragma unroll
    for (int t = 0; t < 32; ++t) s += qW[(h * HDD + t) * DD + d] * aW[t];
    ws[WS_QWEFF + tid] = s;
  } else if (tid < 4096) {                // kWeff with wk = aW[32..63]
    int e = tid - 2048, h = e >> 8, d = e & 255;
    float s = 0.f;
#pragma unroll
    for (int t = 0; t < 32; ++t) s += kW[(h * HDD + t) * DD + d] * aW[HDD + t];
    ws[WS_KWEFF + e] = s;
  } else if (tid < 6144) {                // W2eff with we = aW[64..95]
    int e = tid - 4096, h = e >> 8, c = e & 255;
    float s = 0.f;
#pragma unroll
    for (int t = 0; t < 32; ++t) s += eW2[(h * HDD + t) * DD + c] * aW[2 * HDD + t];
    ws[WS_W2EFF + e] = s;
  } else if (tid < 6168) {                // folded biases
    int e = tid - 6144, g = e >> 3, h = e & 7;
    const float* bv = (g == 0) ? qb : ((g == 1) ? kbias : eb2);
    float s = 0.f;
#pragma unroll
    for (int t = 0; t < 32; ++t) s += bv[h * HDD + t] * aW[g * HDD + t];
    ws[WS_BIAS + e] = s;
  } else if (tid == 6168) {
    ws[WS_BIAS + 24] = ab[0];
  }
}

// ---------------- kernel 1: projections via WMMA -----------------------------
// One wave per 16x16 output tile, K=256 as 8 chained wmma f16 steps.
// Tiles: [0,1024) vproj, [1024,1088) qk, [1088,3136) U/V.
__global__ __launch_bounds__(32) void k1_proj(
    const float* __restrict__ desc, const float* __restrict__ nve,
    const float* __restrict__ vW, const float* __restrict__ vb,
    const float* __restrict__ eW1, float* __restrict__ ws) {
  const int lane = threadIdx.x;
  const int mh = lane & 15;
  const bool hi = lane >= 16;
  const int bt = blockIdx.x;

  const float* Arow;   // lane's A row (M = mh)
  const float* Wrow;   // weight row for output col (N = mh)
  float* Out;
  int ldo, rowbase, Mlim, outcol;
  float bias;

  if (bt < 1024) {                               // vproj = nve @ vW^T + vb
    int mt = bt >> 4, nt = bt & 15;
    rowbase = mt * 16; Mlim = 16;
    outcol = nt * 16 + mh;
    Arow = nve + (size_t)(rowbase + mh) * DD;
    Wrow = vW + (size_t)outcol * DD;
    bias = vb[outcol];
    Out = ws + WS_VPROJ; ldo = DD;
  } else if (bt < 1088) {                        // qk = nve @ [qWeff;kWeff]^T
    int mt = bt - 1024;
    rowbase = mt * 16; Mlim = 16;
    outcol = mh;
    Arow = nve + (size_t)(rowbase + mh) * DD;
    Wrow = (mh < 8) ? ws + WS_QWEFF + (size_t)mh * DD
                    : ws + WS_KWEFF + (size_t)(mh - 8) * DD;
    bias = ws[WS_BIAS + mh];
    Out = ws + WS_QK; ldo = 16;
  } else {                                       // U = desc@W1a^T, V = desc@W1b^T
    int idx = bt - 1088;
    int bb = idx >> 9, rem = idx & 511;
    int mt = rem >> 5, nt = rem & 31;
    rowbase = mt * 16;
    Mlim = SS - rowbase; if (Mlim > 16) Mlim = 16;
    int r = rowbase + mh; if (r >= SS) r = 0;    // clamp read, guarded store
    Arow = desc + ((size_t)bb * SS + r) * DD;
    if (nt < 16) {
      outcol = nt * 16 + mh;
      Wrow = eW1 + (size_t)outcol * (2 * DD);          // W1a row
      Out = ws + WS_U + (size_t)bb * SS1 * DD;
    } else {
      outcol = (nt - 16) * 16 + mh;
      Wrow = eW1 + (size_t)outcol * (2 * DD) + DD;     // W1b row
      Out = ws + WS_V + (size_t)bb * SS1 * DD;
    }
    bias = 0.f; ldo = DD;
  }

  v8f acc = {};
#pragma unroll
  for (int kb2 = 0; kb2 < 8; ++kb2) {
    v16h a, bf;
    const float* ap = Arow + kb2 * 32 + (hi ? 16 : 0);
#pragma unroll
    for (int t = 0; t < 16; ++t) a[t] = (_Float16)ap[t];
    const float* bp = Wrow + kb2 * 32 + (hi ? 16 : 0);
#pragma unroll
    for (int t = 0; t < 16; ++t) bf[t] = (_Float16)bp[t];
    acc = __builtin_amdgcn_wmma_f32_16x16x32_f16(false, a, false, bf, (short)0, acc,
                                                 false, false);
  }
#pragma unroll
  for (int v = 0; v < 8; ++v) {
    int m = v + (hi ? 8 : 0);
    if (m < Mlim) Out[(size_t)(rowbase + m) * ldo + outcol] = acc[v] + bias;
  }
}

// ---------------- kernel 2: fused edge LN + edot WMMA + softmax + attn@v -----
// Block = 128 threads (4 waves) per (b, i) row. Wave w handles j tiles 4w..4w+3.
__global__ __launch_bounds__(128) void k2_edge(
    const float* __restrict__ ws,
    const float* __restrict__ eb1,
    const float* __restrict__ ln_g, const float* __restrict__ ln_b,
    float* __restrict__ out) {
  __shared__ float lgLDS[SS1 * 9];            // [j][head], pad 9 vs banks
  __shared__ _Float16 hstage[4 * 16 * DD];    // per-wave A staging

  const int b = blockIdx.y;
  const int i = blockIdx.x;
  const int tid = threadIdx.x;
  const int w = tid >> 5;
  const int lane = tid & 31;
  const int n = lane & 15;                    // head / M-in-tile role
  const bool hiL = lane >= 16;

  const float* Ub = ws + WS_U + (size_t)b * SS1 * DD;
  const float* Vb = ws + WS_V + (size_t)b * SS1 * DD;
  const float* qk = ws + WS_QK;
  const float* vpj = ws + WS_VPROJ + (size_t)b * SS1 * DD;
  float* ctxOut = out;                                   // (B,S1,H,HD)
  float* attnOut = out + (size_t)BB * SS1 * HH * HDD;    // (B,H,S1,S1)

  // per-lane channel chunk c = lane*8 + q
  float eb1v[8], lngv[8], lnbv[8];
#pragma unroll
  for (int q = 0; q < 8; ++q) {
    eb1v[q] = eb1[lane * 8 + q];
    lngv[q] = ln_g[lane * 8 + q];
    lnbv[q] = ln_b[lane * 8 + q];
  }

  // resident B fragments of W2eff^T (K=256 -> 8 frags); heads >=8 are zero pad
  v16h bw[8];
  {
    int koff = hiL ? 16 : 0;
#pragma unroll
    for (int kb2 = 0; kb2 < 8; ++kb2) {
#pragma unroll
      for (int t = 0; t < 16; ++t)
        bw[kb2][t] = (n < 8) ? (_Float16)ws[WS_W2EFF + n * DD + kb2 * 32 + koff + t]
                             : (_Float16)0.f;
    }
  }

  float rowconst = 0.f;
  if (n < 8)
    rowconst = qk[((size_t)b * SS1 + i) * 16 + n] + ws[WS_BIAS + 16 + n] + ws[WS_BIAS + 24];

  _Float16* hst = hstage + w * (16 * DD);

  for (int t = 0; t < 4; ++t) {
    const int j0 = (w * 4 + t) * 16;
    // --- stage 16 edges: h = U[.] + V[j-1] + eb1, LayerNorm, ReLU, f16 ---
    for (int m = 0; m < 16; ++m) {
      int j = j0 + m;
      int vr = (j == 0) ? 0 : (j - 1);              // dummy row for masked j=0
      int ur = (i == 0) ? vr : (i - 1);             // cls row uses U[j-1]
      const float* up = Ub + (size_t)ur * DD + lane * 8;
      const float* vp = Vb + (size_t)vr * DD + lane * 8;
      float hv[8], s = 0.f, s2 = 0.f;
#pragma unroll
      for (int q = 0; q < 8; ++q) {
        float x = up[q] + vp[q] + eb1v[q];
        hv[q] = x; s += x; s2 += x * x;
      }
      s = wred_sum(s); s2 = wred_sum(s2);
      float mu = s * (1.f / DD);
      float var = s2 * (1.f / DD) - mu * mu;
      float rin = rsqrtf(var + 1e-5f);
      v8h ph;
#pragma unroll
      for (int q = 0; q < 8; ++q) {
        float x = lngv[q] * (hv[q] - mu) * rin + lnbv[q];
        ph[q] = (_Float16)fmaxf(x, 0.f);
      }
      *(v8h*)(hst + m * DD + lane * 8) = ph;
    }
    __syncthreads();

    // --- edot: 16 edges x 8 heads, K=256 via 8 WMMAs ---
    v8f acc = {};
    const int koffA = hiL ? 16 : 0;
#pragma unroll
    for (int kb2 = 0; kb2 < 8; ++kb2) {
      const _Float16* hp = hst + n * DD + kb2 * 32 + koffA;
      v16h a;
#pragma unroll
      for (int q = 0; q < 16; ++q) a[q] = hp[q];
      acc = __builtin_amdgcn_wmma_f32_16x16x32_f16(false, a, false, bw[kb2], (short)0,
                                                   acc, false, false);
    }
    // --- logits: + k-term (+ row consts, softmax-invariant), mask j==0 ---
    if (n < 8) {
#pragma unroll
      for (int v = 0; v < 8; ++v) {
        int m = v + (hiL ? 8 : 0);
        int j = j0 + m;
        float lg = acc[v] + rowconst + qk[((size_t)b * SS1 + j) * 16 + 8 + n];
        if (j == 0) lg = -1e9f;
        lgLDS[j * 9 + n] = lg;
      }
    }
    __syncthreads();
  }

  // --- softmax over j + ctx = attn @ v; wave w owns heads 2w, 2w+1 ---
  for (int hb = 0; hb < 2; ++hb) {
    int hh = w * 2 + hb;
    float lv[8], mx = -3.0e38f;
#pragma unroll
    for (int t3 = 0; t3 < 8; ++t3) {
      lv[t3] = lgLDS[(lane + 32 * t3) * 9 + hh];
      mx = fmaxf(mx, lv[t3]);
    }
    mx = wred_max(mx);
    float s = 0.f;
#pragma unroll
    for (int t3 = 0; t3 < 8; ++t3) { lv[t3] = __expf(lv[t3] - mx); s += lv[t3]; }
    s = wred_sum(s);
    float inv = 1.f / s;
#pragma unroll
    for (int t3 = 0; t3 < 8; ++t3) {
      int j = lane + 32 * t3;
      float p = lv[t3] * inv;
      attnOut[(((size_t)b * HH + hh) * SS1 + i) * SS1 + j] = p;
      lgLDS[j * 9 + hh] = p;                 // stash probs for ctx matvec
    }
    __syncthreads();
    float a2 = 0.f;
    const float* vp = vpj + hh * HDD + lane;
    for (int j = 0; j < SS1; ++j) a2 = fmaf(lgLDS[j * 9 + hh], vp[(size_t)j * DD], a2);
    ctxOut[(((size_t)b * SS1 + i) * HH + hh) * HDD + lane] = a2;
    __syncthreads();
  }
}

// ---------------- launcher ---------------------------------------------------
extern "C" void kernel_launch(void* const* d_in, const int* in_sizes, int n_in,
                              void* d_out, int out_size, void* d_ws, size_t ws_size,
                              hipStream_t stream) {
  const float* desc = (const float*)d_in[0];
  const float* nve  = (const float*)d_in[1];
  const float* qW   = (const float*)d_in[2];
  const float* qb   = (const float*)d_in[3];
  const float* kW   = (const float*)d_in[4];
  const float* kbv  = (const float*)d_in[5];
  const float* vW   = (const float*)d_in[6];
  const float* vb   = (const float*)d_in[7];
  const float* eW1  = (const float*)d_in[8];
  const float* eb1  = (const float*)d_in[9];
  const float* ln_g = (const float*)d_in[10];
  const float* ln_b = (const float*)d_in[11];
  const float* eW2  = (const float*)d_in[12];
  const float* eb2  = (const float*)d_in[13];
  const float* aW   = (const float*)d_in[14];
  const float* ab   = (const float*)d_in[15];
  float* ws = (float*)d_ws;
  float* outF = (float*)d_out;

  (void)in_sizes; (void)n_in; (void)out_size; (void)ws_size;

  k0_precompute<<<25, 256, 0, stream>>>(qW, qb, kW, kbv, eW2, eb2, aW, ab, ws);
  k1_proj<<<3136, 32, 0, stream>>>(desc, nve, vW, vb, eW1, ws);
  k2_edge<<<dim3(SS1, BB), 128, 0, stream>>>(ws, eb1, ln_g, ln_b, outF);
}
